// OPTBlock_687194768156
// MI455X (gfx1250) — matrix operand order repaired and checked by reading, not compiled, not run on hardware
//
#include <hip/hip_runtime.h>
#include <math.h>

// ---------------------------------------------------------------------------
// Problem constants (fixed by the reference)
// ---------------------------------------------------------------------------
constexpr int Ctok = 256;       // channels
constexpr int Ltok = 16384;     // tokens per batch (2^14)
constexpr int Btch = 4;
constexpr int Ntok = Btch * Ltok;   // 65536
constexpr int Hh   = 8;
constexpr int DH   = 32;        // head dim
constexpr int WINW = 256;       // window width
constexpr int NWIN_B = 128;     // windows per batch  ((L-1)/128 + 1)
constexpr int Mwin = Btch * NWIN_B; // 512 windows
constexpr int HID  = 512;
constexpr float EPSf = 1e-5f;

// ---------------------------------------------------------------------------
// WMMA types / helpers (CDNA5 gfx1250, wave32)
// ---------------------------------------------------------------------------
typedef __attribute__((ext_vector_type(16))) __bf16 v16bf;
typedef __attribute__((ext_vector_type(8)))  float  v8f;

__device__ __forceinline__ unsigned short f2bf(float f) {
  unsigned int u = __float_as_uint(f);
  unsigned int r = (u + 0x7FFFu + ((u >> 16) & 1u)) >> 16;   // RNE
  return (unsigned short)r;
}

union FragU { v16bf v; uint4 q[2]; unsigned short s[16]; };

// A-matrix 16x32 bf16 fragment from row-major bf16 tile (stride in elements).
// ISA layout: lane m (0..15) VGPR0-3 hold K=8h..8h+7, VGPR4-7 hold K=16+8h..,
// where h = lane>>4 (lanes 16..31 are the same M rows, upper K interleave).
__device__ __forceinline__ v16bf load_frag_a16(const unsigned short* base,
                                               int stride, int lane) {
  int m = lane & 15, hi = lane >> 4;
  const unsigned short* r = base + m * stride + 8 * hi;
  FragU u;
  u.q[0] = *(const uint4*)(r);
  u.q[1] = *(const uint4*)(r + 16);
  return u.v;
}

// B-matrix 32x16 bf16 fragment where the source is B^T row-major (rows = N
// columns of B, 32 K-elements each). Layout: lane n holds K=8h..8h+7, 16+8h..16+8h+7.
__device__ __forceinline__ v16bf load_frag_bt16(const unsigned short* base,
                                                int stride, int lane) {
  int n = lane & 15, hi = lane >> 4;
  const unsigned short* r = base + n * stride + 8 * hi;
  FragU u;
  u.q[0] = *(const uint4*)(r);
  u.q[1] = *(const uint4*)(r + 16);
  return u.v;
}

// ---- CDNA5 async global->LDS copy (ASYNCcnt), inline asm (toolchain-portable)
__device__ __forceinline__ void async_g2l_b128(const void* gptr, void* lptr) {
  // generic shared-pointer low 32 bits == LDS byte offset (LDS aperture)
  unsigned lds = (unsigned)(size_t)lptr;
  asm volatile("global_load_async_to_lds_b128 %0, %1, off"
               :: "v"(lds), "v"((unsigned long long)(size_t)gptr)
               : "memory");
}
__device__ __forceinline__ void wait_async0() {
  asm volatile("s_wait_asynccnt 0x0" ::: "memory");
}

// block-wide (256 thread) sum reduction
__device__ __forceinline__ float blk_sum(float v, float* red) {
  int c = threadIdx.x;
  red[c] = v; __syncthreads();
  for (int off = 128; off > 0; off >>= 1) {
    if (c < off) red[c] += red[c + off];
    __syncthreads();
  }
  float r = red[0];
  __syncthreads();
  return r;
}

// ---------------------------------------------------------------------------
// 1) per-batch coordinate mean
// ---------------------------------------------------------------------------
__global__ void means_kernel(const float* __restrict__ coord, float* __restrict__ means) {
  int b = blockIdx.x, tid = threadIdx.x;
  float s0 = 0.f, s1 = 0.f, s2 = 0.f;
  for (int l = tid; l < Ltok; l += 256) {
    const float* p = coord + ((size_t)b * Ltok + l) * 3;
    s0 += p[0]; s1 += p[1]; s2 += p[2];
  }
  __shared__ float r0[256], r1[256], r2[256];
  r0[tid] = s0; r1[tid] = s1; r2[tid] = s2; __syncthreads();
  for (int off = 128; off > 0; off >>= 1) {
    if (tid < off) { r0[tid] += r0[tid+off]; r1[tid] += r1[tid+off]; r2[tid] += r2[tid+off]; }
    __syncthreads();
  }
  if (tid == 0) {
    means[b*3+0] = r0[0] / (float)Ltok;
    means[b*3+1] = r1[0] / (float)Ltok;
    means[b*3+2] = r2[0] / (float)Ltok;
  }
}

// ---------------------------------------------------------------------------
// 2) pe = gelu(LN(xyz_c @ lin_w^T + b));  inp = feat + pe
// ---------------------------------------------------------------------------
__global__ void pe_inp_kernel(const float* __restrict__ feat, const float* __restrict__ coord,
                              const float* __restrict__ means,
                              const float* __restrict__ lin_w, const float* __restrict__ lin_b,
                              const float* __restrict__ ln_g, const float* __restrict__ ln_b,
                              float* __restrict__ inp) {
  int t = blockIdx.x, c = threadIdx.x;
  int b = t >> 14;
  float x0 = coord[(size_t)t*3+0] - means[b*3+0];
  float x1 = coord[(size_t)t*3+1] - means[b*3+1];
  float x2 = coord[(size_t)t*3+2] - means[b*3+2];
  float v = x0*lin_w[c*3+0] + x1*lin_w[c*3+1] + x2*lin_w[c*3+2] + lin_b[c];
  __shared__ float red[256];
  float mu  = blk_sum(v, red) * (1.f/256.f);
  float d   = v - mu;
  float var = blk_sum(d*d, red) * (1.f/256.f);
  float y   = d * rsqrtf(var + EPSf) * ln_g[c] + ln_b[c];
  float gl  = 0.5f * y * (1.f + erff(y * 0.70710678118654752f));   // exact gelu
  inp[(size_t)t*Ctok + c] = feat[(size_t)t*Ctok + c] + gl;
}

// ---------------------------------------------------------------------------
// 3) depthwise conv(k=3,SAME) + residual + LN -> feat1 ; LN(feat1) -> x_in(bf16)
// ---------------------------------------------------------------------------
__global__ void conv_feat1_kernel(const float* __restrict__ feat, const float* __restrict__ inp,
                                  const float* __restrict__ conv_w, const float* __restrict__ conv_b,
                                  const float* __restrict__ g1, const float* __restrict__ b1,
                                  const float* __restrict__ g2, const float* __restrict__ b2,
                                  float* __restrict__ feat1, unsigned short* __restrict__ xin_bf) {
  int t = blockIdx.x, c = threadIdx.x;
  int l = t & (Ltok - 1);
  float cm = (l > 0)          ? inp[(size_t)(t-1)*Ctok + c] : 0.f;
  float cc =                    inp[(size_t)t    *Ctok + c];
  float cp = (l < Ltok - 1)   ? inp[(size_t)(t+1)*Ctok + c] : 0.f;
  float conv = conv_w[0*Ctok+c]*cm + conv_w[1*Ctok+c]*cc + conv_w[2*Ctok+c]*cp;
  float v = feat[(size_t)t*Ctok + c] + conv + conv_b[c];
  __shared__ float red[256];
  float mu  = blk_sum(v, red) * (1.f/256.f);
  float d   = v - mu;
  float var = blk_sum(d*d, red) * (1.f/256.f);
  float f1  = d * rsqrtf(var + EPSf) * g1[c] + b1[c];
  feat1[(size_t)t*Ctok + c] = f1;
  float mu2  = blk_sum(f1, red) * (1.f/256.f);
  float d2   = f1 - mu2;
  float var2 = blk_sum(d2*d2, red) * (1.f/256.f);
  float xi   = d2 * rsqrtf(var2 + EPSf) * g2[c] + b2[c];
  xin_bf[(size_t)t*Ctok + c] = f2bf(xi);
}

// ---------------------------------------------------------------------------
// f32 -> bf16 conversion (weights)
// ---------------------------------------------------------------------------
__global__ void cvt_bf16_kernel(const float* __restrict__ src, unsigned short* __restrict__ dst, int n) {
  int i = blockIdx.x * 256 + threadIdx.x;
  if (i < n) dst[i] = f2bf(src[i]);
}

// ---------------------------------------------------------------------------
// Generic WMMA GEMM:  C[M,N] = A[M,K](bf16) x B^T[N,K](bf16)
// block tile 128x64, 8 waves, wave w owns row-tile w and 4 col-tiles.
// Double-buffered LDS staging via CDNA5 async global->LDS copies (ASYNCcnt):
// buffer p^1 is filled asynchronously while WMMAs consume buffer p.
// mode 0: outBF = C                      (qkv)
// mode 1: outF  = C                      (proj, raw; bias folded in later)
// mode 2: outBF = relu((C+bias)*bn_s+bn_b)  (ffn1 + "batchnorm")
// mode 3: accum += C + bias              (ffn2 residual into d_out)
// ---------------------------------------------------------------------------
__global__ void wmma_gemm_kernel(const unsigned short* __restrict__ A,
                                 const unsigned short* __restrict__ Bt,
                                 int Ncols, int K, int mode,
                                 float* __restrict__ outF, unsigned short* __restrict__ outBF,
                                 float* __restrict__ accum,
                                 const float* __restrict__ bias,
                                 const float* __restrict__ bn_g, const float* __restrict__ bn_b) {
  __shared__ __align__(16) unsigned short As[2][128 * 32];   // 2 x 8KB
  __shared__ __align__(16) unsigned short Bs[2][64 * 32];    // 2 x 4KB
  const int rowBase = blockIdx.y * 128;
  const int colBase = blockIdx.x * 64;
  const int tid = threadIdx.x;
  const int wid = tid >> 5, lane = tid & 31;

  // Issue async fills for one K-tile into buffer `buf`.
  // A tile: 128 rows x 64B  -> 512 b128 slots, 2 per thread.
  // B tile:  64 rows x 64B  -> 256 b128 slots, 1 per thread.
  auto issue_tile = [&](int buf, int k0) {
    #pragma unroll
    for (int j = 0; j < 2; ++j) {
      int s = j * 256 + tid;          // 0..511
      int r = s >> 2, q = s & 3;      // row, 16B-quarter
      async_g2l_b128(A + (size_t)(rowBase + r) * K + k0 + 8 * q,
                     &As[buf][r * 32 + 8 * q]);
    }
    {
      int r = tid >> 2, q = tid & 3;  // row 0..63, quarter
      async_g2l_b128(Bt + (size_t)(colBase + r) * K + k0 + 8 * q,
                     &Bs[buf][r * 32 + 8 * q]);
    }
  };

  v8f acc[4];
  for (int t = 0; t < 4; ++t)
    for (int i = 0; i < 8; ++i) acc[t][i] = 0.f;

  issue_tile(0, 0);
  int parity = 0;
  for (int k0 = 0; k0 < K; k0 += 32) {
    wait_async0();        // this wave's async fills of buf[parity] are done
    __syncthreads();      // => all waves' fills done; prev compute done too
    if (k0 + 32 < K) issue_tile(parity ^ 1, k0 + 32);   // overlap with WMMAs
    v16bf a = load_frag_a16(&As[parity][wid * 16 * 32], 32, lane);
    #pragma unroll
    for (int ct = 0; ct < 4; ++ct) {
      v16bf b = load_frag_bt16(&Bs[parity][ct * 16 * 32], 32, lane);
      acc[ct] = __builtin_amdgcn_wmma_f32_16x16x32_bf16(
          false, a, false, b, (short)0, acc[ct], false, false);
    }
    parity ^= 1;
  }

  const int n = lane & 15, hi = lane >> 4;
  for (int ct = 0; ct < 4; ++ct) {
    int col = colBase + ct * 16 + n;
    for (int g = 0; g < 8; ++g) {
      int row = rowBase + wid * 16 + g + 8 * hi;   // D layout: M = g + 8*(lane>>4)
      size_t o = (size_t)row * Ncols + col;
      float v = acc[ct][g];
      if (mode == 0) {
        outBF[o] = f2bf(v);
      } else if (mode == 1) {
        outF[o] = v;
      } else if (mode == 2) {
        float s = bn_g[col] * rsqrtf(1.f + EPSf);
        float t = (v + bias[col]) * s + bn_b[col];
        outBF[o] = f2bf(fmaxf(t, 0.f));
      } else {
        accum[o] += v + bias[col];
      }
    }
  }
}

// ---------------------------------------------------------------------------
// Attention: one block = (window m, head h, 64-query slab qb). 8 waves.
// Gathered Q/K/V rows copied global->LDS with async b128 copies.
// S = QK^T/sqrt(dh) masked -> LDS f32; row softmax (keep unnormalized exp);
// out = (P V) / rowsum via WMMA, written bf16 to attn_out[(m*256+q)*256 + h*32+d]
// ---------------------------------------------------------------------------
constexpr int ATTN_SMEM =
    64*256*4      /* S      */
  + 64*32*2       /* Qs     */
  + 256*32*2      /* Ks     */
  + 256*32*2      /* Vs     */
  + 64*4          /* sums   */
  + 256*4;        /* maskf  */   // = 103680 bytes (< 320KB WGP LDS)

__global__ void attn_kernel(const unsigned short* __restrict__ qkv,
                            const int* __restrict__ win_idx,
                            const unsigned char* __restrict__ win_mask,
                            unsigned short* __restrict__ attn_out) {
  extern __shared__ char smem[];
  float*          S     = (float*)(smem);                       // [64][256]
  unsigned short* Qs    = (unsigned short*)(smem + 65536);      // [64][32]
  unsigned short* Ks    = (unsigned short*)(smem + 69632);      // [256][32]
  unsigned short* Vs    = (unsigned short*)(smem + 86016);      // [256][32]
  float*          sums  = (float*)(smem + 102400);              // [64]
  float*          maskf = (float*)(smem + 102656);              // [256]

  const int qb = blockIdx.x, h = blockIdx.y, m = blockIdx.z;
  const int tid = threadIdx.x;
  const int wid = tid >> 5, lane = tid & 31;

  // ---- async gather of K, V (all 256 rows), Q (64 rows); mask to LDS ------
  {
    int row = tid;
    int tok = win_idx[m * WINW + row];
    const unsigned short* kg = qkv + (size_t)tok * 768 + 256 + h * DH;
    const unsigned short* vg = qkv + (size_t)tok * 768 + 512 + h * DH;
    #pragma unroll
    for (int j = 0; j < 4; ++j) {
      async_g2l_b128(kg + 8 * j, Ks + row * DH + 8 * j);
      async_g2l_b128(vg + 8 * j, Vs + row * DH + 8 * j);
    }
    maskf[row] = win_mask[m * WINW + row] ? 1.f : 0.f;
  }
  if (tid < 64) {
    int row = qb * 64 + tid;
    int tok = win_idx[m * WINW + row];
    const unsigned short* qg = qkv + (size_t)tok * 768 + 0 + h * DH;
    #pragma unroll
    for (int j = 0; j < 4; ++j)
      async_g2l_b128(qg + 8 * j, Qs + tid * DH + 8 * j);
  }
  wait_async0();
  __syncthreads();

  // ---- scores: 4 row-tiles x 16 col-tiles, K=32 in one wmma ---------------
  const float scale = 0.17677669529663687f;   // 1/sqrt(32)
  for (int t = wid; t < 64; t += 8) {
    int rt = t >> 4, ct = t & 15;
    v16bf a = load_frag_a16(Qs + rt * 16 * DH, DH, lane);
    v16bf b = load_frag_bt16(Ks + ct * 16 * DH, DH, lane);
    v8f c;
    for (int i = 0; i < 8; ++i) c[i] = 0.f;
    c = __builtin_amdgcn_wmma_f32_16x16x32_bf16(false, a, false, b, (short)0, c, false, false);
    int n = lane & 15, hi = lane >> 4;
    for (int g = 0; g < 8; ++g) {
      int row = rt * 16 + g + 8 * hi;
      int col = ct * 16 + n;
      S[row * 256 + col] = (maskf[col] > 0.5f) ? c[g] * scale : -1.0e9f;
    }
  }
  __syncthreads();

  // ---- row softmax (store unnormalized exp, keep row sums) ----------------
  if (tid < 64) {
    float mx = -3.4e38f;
    const int base = tid * 256;
    for (int c0 = 0; c0 < 256; ++c0) mx = fmaxf(mx, S[base + c0]);
    float sm = 0.f;
    for (int c0 = 0; c0 < 256; ++c0) {
      float e = __expf(S[base + c0] - mx);
      S[base + c0] = e;
      sm += e;
    }
    sums[tid] = sm;
  }
  __syncthreads();

  // ---- out = P V : 4 row-tiles x 2 d-tiles = 8 tiles, one per wave --------
  {
    int rt = wid >> 1, dt = wid & 1;
    v8f acc;
    for (int i = 0; i < 8; ++i) acc[i] = 0.f;
    int n = lane & 15, hi = lane >> 4;
    for (int k0 = 0; k0 < 256; k0 += 32) {
      FragU ua;                     // A-frag of P: f32 LDS -> bf16 on the fly
      const float* srow = S + (rt * 16 + n) * 256 + k0;
      for (int j = 0; j < 8; ++j)  ua.s[j]     = f2bf(srow[8 * hi + j]);
      for (int j = 0; j < 8; ++j)  ua.s[8 + j] = f2bf(srow[16 + 8 * hi + j]);
      FragU ub;                     // B-frag: column d of V over K slab
      int d = dt * 16 + n;
      for (int j = 0; j < 16; ++j) ub.s[j] = Vs[(k0 + 8 * hi + j + (j >= 8 ? 8 : 0)) * DH + d];
      acc = __builtin_amdgcn_wmma_f32_16x16x32_bf16(
          false, ua.v, false, ub.v, (short)0, acc, false, false);
    }
    for (int g = 0; g < 8; ++g) {
      int row = rt * 16 + g + 8 * hi;
      int qr  = qb * 64 + row;
      float v = acc[g] / sums[row];
      attn_out[((size_t)(m * WINW + qr)) * Ctok + h * DH + dt * 16 + n] = f2bf(v);
    }
  }
}

// ---------------------------------------------------------------------------
// 7) combine: x_out = feat1 + gather(proj_out)/count + proj_b ; x2 = LN(x_out)
//    token l -> (win l/128, row l%128) and, if l>=128, (win l/128-1, row l%128+128)
// ---------------------------------------------------------------------------
__global__ void combine_kernel(const float* __restrict__ feat1, const float* __restrict__ proj_out,
                               const float* __restrict__ proj_b,
                               const float* __restrict__ g, const float* __restrict__ bb,
                               float* __restrict__ xout, unsigned short* __restrict__ x2bf) {
  int t = blockIdx.x, c = threadIdx.x;
  int l = t & (Ltok - 1);
  int b = t >> 14;
  int w2 = l >> 7, r2 = l & 127;
  size_t rowA = ((size_t)(b * NWIN_B + w2)) * WINW + r2;
  float v = proj_out[rowA * Ctok + c];
  float cnt = 1.f;
  if (l >= 128) {
    size_t rowB = ((size_t)(b * NWIN_B + w2 - 1)) * WINW + (r2 + 128);
    v += proj_out[rowB * Ctok + c];
    cnt = 2.f;
  }
  float xo = feat1[(size_t)t * Ctok + c] + v / cnt + proj_b[c];
  xout[(size_t)t * Ctok + c] = xo;

  __shared__ float red[256];
  float mu  = blk_sum(xo, red) * (1.f/256.f);
  float d   = xo - mu;
  float var = blk_sum(d*d, red) * (1.f/256.f);
  float x2  = d * rsqrtf(var + EPSf) * g[c] + bb[c];
  x2bf[(size_t)t * Ctok + c] = f2bf(x2);
}

// ---------------------------------------------------------------------------
// Host launch
// ---------------------------------------------------------------------------
extern "C" void kernel_launch(void* const* d_in, const int* in_sizes, int n_in,
                              void* d_out, int out_size, void* d_ws, size_t ws_size,
                              hipStream_t stream) {
  (void)in_sizes; (void)n_in; (void)out_size; (void)ws_size;
  const float* feat     = (const float*)d_in[0];
  const float* coord    = (const float*)d_in[1];
  const int*   win_idx  = (const int*)d_in[3];
  const unsigned char* win_mask = (const unsigned char*)d_in[4];
  const float* cpe_lin_w  = (const float*)d_in[5];
  const float* cpe_lin_b  = (const float*)d_in[6];
  const float* cpe_ln_g   = (const float*)d_in[7];
  const float* cpe_ln_b   = (const float*)d_in[8];
  const float* cpe_conv_w = (const float*)d_in[9];
  const float* cpe_conv_b = (const float*)d_in[10];
  const float* cpe_norm_g = (const float*)d_in[11];
  const float* cpe_norm_b = (const float*)d_in[12];
  const float* norm1_g    = (const float*)d_in[13];
  const float* norm1_b    = (const float*)d_in[14];
  const float* qkv_w      = (const float*)d_in[15];
  const float* proj_w     = (const float*)d_in[16];
  const float* proj_b     = (const float*)d_in[17];
  const float* norm2_g    = (const float*)d_in[18];
  const float* norm2_b    = (const float*)d_in[19];
  const float* ffn_w1     = (const float*)d_in[20];
  const float* ffn_b1     = (const float*)d_in[21];
  const float* bn_g       = (const float*)d_in[22];
  const float* bn_b       = (const float*)d_in[23];
  const float* ffn_w2     = (const float*)d_in[24];
  const float* ffn_b2     = (const float*)d_in[25];

  char* ws = (char*)d_ws;
  constexpr size_t MB = 1u << 20;
  // small region: means + bf16 weights
  float*          means    = (float*)(ws + 0);
  unsigned short* wqkv_bf  = (unsigned short*)(ws + 1024);
  unsigned short* wproj_bf = wqkv_bf  + 768 * 256;
  unsigned short* wf1_bf   = wproj_bf + 256 * 256;
  unsigned short* wf2_bf   = wf1_bf   + 512 * 256;
  // Region A (128MB): inp f32 + x_in bf16 early; proj_out f32 later (both dead)
  float*          inp      = (float*)(ws + 2 * MB);                  // 64MB
  unsigned short* xin_bf   = (unsigned short*)(ws + 2 * MB + 64 * MB); // 32MB
  float*          proj_out = (float*)(ws + 2 * MB);                  // 128MB
  // Region B (96MB): qkv bf16; later x2 bf16 (32MB) + h1 bf16 (64MB)
  unsigned short* qkv_bf   = (unsigned short*)(ws + 130 * MB);       // 96MB
  unsigned short* x2_bf    = (unsigned short*)(ws + 130 * MB);       // 32MB
  unsigned short* h1_bf    = (unsigned short*)(ws + 130 * MB + 32 * MB); // 64MB
  // Region C/D
  float*          feat1    = (float*)(ws + 226 * MB);                // 64MB
  unsigned short* aout_bf  = (unsigned short*)(ws + 290 * MB);       // 64MB

  float* xout = (float*)d_out;

  // 1) coord means
  means_kernel<<<Btch, 256, 0, stream>>>(coord, means);
  // 2) conditional positional encoding -> inp
  pe_inp_kernel<<<Ntok, 256, 0, stream>>>(feat, coord, means, cpe_lin_w, cpe_lin_b,
                                          cpe_ln_g, cpe_ln_b, inp);
  // 3) depthwise conv + LN -> feat1, LN -> x_in (bf16)
  conv_feat1_kernel<<<Ntok, 256, 0, stream>>>(feat, inp, cpe_conv_w, cpe_conv_b,
                                              cpe_norm_g, cpe_norm_b, norm1_g, norm1_b,
                                              feat1, xin_bf);
  // 4) weights to bf16
  cvt_bf16_kernel<<<(768*256+255)/256, 256, 0, stream>>>(qkv_w,  wqkv_bf,  768*256);
  cvt_bf16_kernel<<<(256*256+255)/256, 256, 0, stream>>>(proj_w, wproj_bf, 256*256);
  cvt_bf16_kernel<<<(512*256+255)/256, 256, 0, stream>>>(ffn_w1, wf1_bf,   512*256);
  cvt_bf16_kernel<<<(256*512+255)/256, 256, 0, stream>>>(ffn_w2, wf2_bf,   256*512);
  // 5) qkv = x_in @ qkv_w^T   (per token, 65536 x 768 x 256)
  wmma_gemm_kernel<<<dim3(768/64, Ntok/128), 256, 0, stream>>>(
      xin_bf, wqkv_bf, 768, 256, /*mode=*/0, nullptr, qkv_bf, nullptr,
      nullptr, nullptr, nullptr);
  // 6) windowed attention
  attn_kernel<<<dim3(WINW/64, Hh, Mwin), 256, ATTN_SMEM, stream>>>(
      qkv_bf, win_idx, win_mask, aout_bf);
  // 7) proj GEMM (131072 x 256 x 256) -> f32 (bias folded into combine)
  wmma_gemm_kernel<<<dim3(256/64, (Mwin*WINW)/128), 256, 0, stream>>>(
      aout_bf, wproj_bf, 256, 256, /*mode=*/1, proj_out, nullptr, nullptr,
      nullptr, nullptr, nullptr);
  // 8) gather-combine + residual -> x_out (d_out); LN -> x2 (bf16)
  combine_kernel<<<Ntok, 256, 0, stream>>>(feat1, proj_out, proj_b,
                                           norm2_g, norm2_b, xout, x2_bf);
  // 9) ffn1: relu((x2 @ w1^T + b1) * bn_g/sqrt(1+eps) + bn_b) -> h1 (bf16)
  wmma_gemm_kernel<<<dim3(HID/64, Ntok/128), 256, 0, stream>>>(
      x2_bf, wf1_bf, HID, 256, /*mode=*/2, nullptr, h1_bf, nullptr,
      ffn_b1, bn_g, bn_b);
  // 10) ffn2: d_out += h1 @ w2^T + b2
  wmma_gemm_kernel<<<dim3(256/64, Ntok/128), 256, 0, stream>>>(
      h1_bf, wf2_bf, 256, HID, /*mode=*/3, nullptr, nullptr, xout,
      ffn_b2, nullptr, nullptr);
}
